// HeteroGATv2_2662879723778
// MI455X (gfx1250) — compile-verified, matrix operand order checked by reading, not run on hardware
//
#include <hip/hip_runtime.h>
#include <hip/hip_bf16.h>

typedef __attribute__((ext_vector_type(16))) _Float16 v16h;
typedef __attribute__((ext_vector_type(8)))  _Float16 v8h;
typedef __attribute__((ext_vector_type(8)))  float    v8f;

#define N_WORD 20000
#define N_DOC  20000
#define NE     500000
#define DIM    128

// ---------------------------------------------------------------------------
// WMMA GEMM:  C[M,N] = act(A[M,K] @ B + bias[N]),  B pre-transposed f16 [N,Kp]
// One wave -> 16x32 output tile (one A fragment feeds two WMMAs).
// Preconditions: M%16==0 live tiles, N%32==0, K%32==0, 16B-aligned rows.
// A frag (16x32 f16): lane m=lane&15 is row; kb=8*(lane>>4);
//   halves 0..7 -> K=k+kb+0..7 ; halves 8..15 -> K=k+16+kb+0..7
// B frag: same K mapping, lane&15 is the column (contiguous in Bt row).
// D (8 f32/lane): elem r -> row=row0+r+8*(lane>>4), col=col0+(lane&15).
// ---------------------------------------------------------------------------
__global__ __launch_bounds__(128)
void gemm_wmma(const float* __restrict__ A, const _Float16* __restrict__ Bt,
               const float* __restrict__ bias, float* __restrict__ C,
               int M, int N, int K, int lda, int ldb, int act)
{
    const int lane  = threadIdx.x & 31;
    const int wave  = threadIdx.x >> 5;
    const int tileM = blockIdx.y * 4 + wave;
    const int row0  = tileM * 16;
    if (row0 >= M) return;                          // wave-uniform
    const int col0  = blockIdx.x * 32;

    const int m  = lane & 15;
    const int kb = (lane >> 4) * 8;                 // 0 or 8

    const float*    Ar = A  + (long)(row0 + m) * lda + kb;
    const _Float16* B0 = Bt + (long)(col0 + m) * ldb + kb;
    const _Float16* B1 = B0 + (long)16 * ldb;

    v8f acc0 = {}, acc1 = {};
    for (int k = 0; k < K; k += 32) {
        const float4 a0 = *(const float4*)(Ar + k);
        const float4 a1 = *(const float4*)(Ar + k + 4);
        const float4 a2 = *(const float4*)(Ar + k + 16);
        const float4 a3 = *(const float4*)(Ar + k + 20);
        v16h a;
        a[0]=(_Float16)a0.x; a[1]=(_Float16)a0.y; a[2]=(_Float16)a0.z; a[3]=(_Float16)a0.w;
        a[4]=(_Float16)a1.x; a[5]=(_Float16)a1.y; a[6]=(_Float16)a1.z; a[7]=(_Float16)a1.w;
        a[8]=(_Float16)a2.x; a[9]=(_Float16)a2.y; a[10]=(_Float16)a2.z; a[11]=(_Float16)a2.w;
        a[12]=(_Float16)a3.x; a[13]=(_Float16)a3.y; a[14]=(_Float16)a3.z; a[15]=(_Float16)a3.w;

        const v8h b0lo = *(const v8h*)(B0 + k);
        const v8h b0hi = *(const v8h*)(B0 + k + 16);
        const v8h b1lo = *(const v8h*)(B1 + k);
        const v8h b1hi = *(const v8h*)(B1 + k + 16);
        v16h b0, b1;
#pragma unroll
        for (int j = 0; j < 8; ++j) {
            b0[j] = b0lo[j]; b0[j + 8] = b0hi[j];
            b1[j] = b1lo[j]; b1[j + 8] = b1hi[j];
        }
        acc0 = __builtin_amdgcn_wmma_f32_16x16x32_f16(false, a, false, b0,
                                                      (short)0, acc0, false, false);
        acc1 = __builtin_amdgcn_wmma_f32_16x16x32_f16(false, a, false, b1,
                                                      (short)0, acc1, false, false);
    }

    const int n0  = col0 + (lane & 15);
    const int mhi = (lane >> 4) * 8;
    const float bv0 = bias ? bias[n0]      : 0.f;
    const float bv1 = bias ? bias[n0 + 16] : 0.f;
#pragma unroll
    for (int r = 0; r < 8; ++r) {
        const long rr = (long)(row0 + mhi + r);
        float v0 = acc0[r] + bv0;
        float v1 = acc1[r] + bv1;
        if (act) { v0 = fmaxf(v0, 0.f); v1 = fmaxf(v1, 0.f); }
        C[rr * N + n0]      = v0;
        C[rr * N + n0 + 16] = v1;
    }
}

// ---------------------------------------------------------------------------
// One-time operand prep: weight f32[K,N] -> f16 transposed [N,Kp] (zero pad),
// and x_word -> K-padded f32 copy.
// ---------------------------------------------------------------------------
__global__ void w_to_f16t(const float* __restrict__ W, _Float16* __restrict__ Wt,
                          int K, int N, int Kp)
{
    const long i = (long)blockIdx.x * blockDim.x + threadIdx.x;
    if (i >= (long)N * Kp) return;
    const int n = (int)(i / Kp), k = (int)(i % Kp);
    Wt[i] = (k < K) ? (_Float16)W[(long)k * N + n] : (_Float16)0.f;
}

__global__ void pad_copy(const float* __restrict__ X, float* __restrict__ Xp,
                         int K, int Kp, long n)
{
    const long i = (long)blockIdx.x * blockDim.x + threadIdx.x;
    if (i >= n) return;
    const int k = (int)(i % Kp);
    const long m = i / Kp;
    Xp[i] = (k < K) ? X[m * K + k] : 0.f;
}

// ---------------------------------------------------------------------------
// Edge kernels (wave-per-edge, coalesced 512B node-feature rows from L2)
// ---------------------------------------------------------------------------
__device__ inline void atomicMaxFloat(float* addr, float val)
{
    unsigned int* ua  = (unsigned int*)addr;
    unsigned int  old = *ua;
    while (__uint_as_float(old) < val) {
        unsigned int assumed = old;
        old = atomicCAS(ua, assumed, __float_as_uint(val));
        if (old == assumed) break;
    }
}

__global__ __launch_bounds__(256)
void edge_score_max(const float* __restrict__ fs, const float* __restrict__ fd,
                    const float* __restrict__ a,  const int* __restrict__ src,
                    const int* __restrict__ dst,  float* __restrict__ score,
                    float* __restrict__ segmax, int nE)
{
    const int wid  = blockIdx.x * (blockDim.x >> 5) + (threadIdx.x >> 5);
    const int lane = threadIdx.x & 31;
    if (wid >= nE) return;
    const int s = src[wid], d = dst[wid];
    const float* fsr = fs + (long)s * DIM;
    const float* fdr = fd + (long)d * DIM;
#pragma unroll
    for (int h = 0; h < 4; ++h) {
        float x = fsr[h * 32 + lane] + fdr[h * 32 + lane];
        x = (x > 0.f) ? x : 0.2f * x;               // leaky_relu(0.2)
        float p = x * a[h * 32 + lane];
#pragma unroll
        for (int off = 16; off > 0; off >>= 1)
            p += __shfl_xor(p, off, 32);
        if (lane == 0) {
            score[(long)wid * 4 + h] = p;
            atomicMaxFloat(&segmax[(long)d * 4 + h], p);
        }
    }
}

__global__ __launch_bounds__(256)
void edge_exp_sum(const float* __restrict__ segmax, const int* __restrict__ dst,
                  float* __restrict__ score, float* __restrict__ segsum, long EH)
{
    const long i = (long)blockIdx.x * blockDim.x + threadIdx.x;
    if (i >= EH) return;
    const int e = (int)(i >> 2), h = (int)(i & 3);
    const int d = dst[e];
    const float ex = __expf(score[i] - segmax[(long)d * 4 + h]);
    score[i] = ex;
    atomicAdd(&segsum[(long)d * 4 + h], ex);
}

__global__ __launch_bounds__(256)
void edge_aggregate(const float* __restrict__ fs, const float* __restrict__ ex,
                    const float* __restrict__ segsum, const int* __restrict__ src,
                    const int* __restrict__ dst, float* __restrict__ out, int nE)
{
    const int wid  = blockIdx.x * (blockDim.x >> 5) + (threadIdx.x >> 5);
    const int lane = threadIdx.x & 31;
    if (wid >= nE) return;
    const int s = src[wid], d = dst[wid];
    const int h = lane >> 3;                        // 4 consecutive feats, one head
    const float alpha = ex[(long)wid * 4 + h] / segsum[(long)d * 4 + h];
    const float4 f = *(const float4*)(fs + (long)s * DIM + lane * 4);
    float* o = out + (long)d * DIM + lane * 4;
    atomicAdd(o + 0, alpha * f.x);
    atomicAdd(o + 1, alpha * f.y);
    atomicAdd(o + 2, alpha * f.z);
    atomicAdd(o + 3, alpha * f.w);
}

// ---------------------------------------------------------------------------
// Init / combine helpers
// ---------------------------------------------------------------------------
__global__ void fill_f32(float* __restrict__ p, float v, long n)
{
    const long i = (long)blockIdx.x * blockDim.x + threadIdx.x;
    if (i < n) p[i] = v;
}
__global__ void init_bias128(float* __restrict__ p, const float* __restrict__ b, long n)
{
    const long i = (long)blockIdx.x * blockDim.x + threadIdx.x;
    if (i < n) p[i] = b[(int)(i & (DIM - 1))];
}
__global__ void combine_mean2(float* __restrict__ o, const float* __restrict__ a,
                              const float* __restrict__ b, long n)
{
    const long i = (long)blockIdx.x * blockDim.x + threadIdx.x;
    if (i < n) o[i] = 0.5f * (a[i] + b[i]);
}

// ---------------------------------------------------------------------------
extern "C" void kernel_launch(void* const* d_in, const int* in_sizes, int n_in,
                              void* d_out, int out_size, void* d_ws, size_t ws_size,
                              hipStream_t stream)
{
    (void)in_sizes; (void)n_in; (void)out_size; (void)ws_size;

    const float* x_word  = (const float*)d_in[0];
    const float* x_doc   = (const float*)d_in[1];
    const float* Wp_word = (const float*)d_in[2];
    const float* bp_word = (const float*)d_in[3];
    const float* Wp_doc  = (const float*)d_in[4];
    const float* bp_doc  = (const float*)d_in[5];
    const float* Wsrc    = (const float*)d_in[6];
    const float* bsrc    = (const float*)d_in[7];
    const float* Wdst    = (const float*)d_in[8];
    const float* bdst    = (const float*)d_in[9];
    const float* attn    = (const float*)d_in[10];
    const float* bconv   = (const float*)d_in[11];
    const float* Wc1     = (const float*)d_in[12];
    const float* bc1     = (const float*)d_in[13];
    const float* Wc2     = (const float*)d_in[14];
    const float* bc2     = (const float*)d_in[15];
    const int*   src_wd  = (const int*)d_in[16];
    const int*   dst_wd  = (const int*)d_in[17];
    const int*   src_dw  = (const int*)d_in[18];
    const int*   dst_dw  = (const int*)d_in[19];
    const int*   src_dd  = (const int*)d_in[20];
    const int*   dst_dd  = (const int*)d_in[21];

    // ---- carve workspace ----
    float* ws = (float*)d_ws;
    const long ND_D = (long)N_DOC * DIM;            // 2.56M floats
    float* hw     = ws;  ws += ND_D;
    float* hd     = ws;  ws += ND_D;
    float* fs     = ws;  ws += ND_D;
    float* fd     = ws;  ws += ND_D;
    float* owd    = ws;  ws += ND_D;
    float* odw    = ws;  ws += ND_D;
    float* odd    = ws;  ws += ND_D;
    float* score  = ws;  ws += (long)NE * 4;
    float* segmax = ws;  ws += (long)N_DOC * 4;
    float* segsum = ws;  ws += (long)N_DOC * 4;
    float* xp     = ws;  ws += (long)N_WORD * 320;  // K=300 padded to 320
    float* z      = score;                           // MLP hidden reuses score

    // f16 transposed weights
    _Float16* wh = (_Float16*)ws;
    _Float16* WpW_t = wh;  wh += (long)DIM * 320;   // [128,320]
    _Float16* WpD_t = wh;  wh += (long)DIM * DIM;   // [128,128]
    _Float16* Wsrc_t[6], *Wdst_t[6];
    for (int r = 0; r < 6; ++r) { Wsrc_t[r] = wh; wh += (long)DIM * DIM; }
    for (int r = 0; r < 6; ++r) { Wdst_t[r] = wh; wh += (long)DIM * DIM; }
    _Float16* Wc1_t = wh;  wh += (long)64 * DIM;    // [64,128]
    _Float16* Wc2_t = wh;  wh += (long)64 * 64;     // [64,64]

    auto w16 = [&](const float* W, _Float16* Wt, int K, int N, int Kp) {
        const long n = (long)N * Kp;
        w_to_f16t<<<(int)((n + 255) / 256), 256, 0, stream>>>(W, Wt, K, N, Kp);
    };

    // ---- operand prep (tiny) ----
    w16(Wp_word, WpW_t, 300, DIM, 320);
    w16(Wp_doc,  WpD_t, DIM, DIM, DIM);
    for (int r = 0; r < 6; ++r) {
        w16(Wsrc + (long)r * DIM * DIM, Wsrc_t[r], DIM, DIM, DIM);
        w16(Wdst + (long)r * DIM * DIM, Wdst_t[r], DIM, DIM, DIM);
    }
    w16(Wc1, Wc1_t, DIM, 64, DIM);
    w16(Wc2, Wc2_t, 64,  64, 64);
    {
        const long n = (long)N_WORD * 320;
        pad_copy<<<(int)((n + 255) / 256), 256, 0, stream>>>(x_word, xp, 300, 320, n);
    }

    auto gemm = [&](const float* A, const _Float16* Bt, const float* b, float* C,
                    int M, int N, int K, int lda, int act) {
        dim3 grid(N / 32, (M + 63) / 64);
        gemm_wmma<<<grid, 128, 0, stream>>>(A, Bt, b, C, M, N, K, lda, K, act);
    };

    auto gat = [&](const float* hs, int Ns, const float* hdf, int Nd,
                   const _Float16* Wsp, const float* bsp, const _Float16* Wdp,
                   const float* bdp, const float* a, const float* bias,
                   const int* src, const int* dst, float* out) {
        gemm(hs,  Wsp, bsp, fs, Ns, DIM, DIM, DIM, 0);
        gemm(hdf, Wdp, bdp, fd, Nd, DIM, DIM, DIM, 0);
        const long nd4 = (long)Nd * 4;
        fill_f32<<<(int)((nd4 + 255) / 256), 256, 0, stream>>>(segmax, -3.0e38f, nd4);
        fill_f32<<<(int)((nd4 + 255) / 256), 256, 0, stream>>>(segsum, 0.f, nd4);
        const long ndD = (long)Nd * DIM;
        init_bias128<<<(int)((ndD + 255) / 256), 256, 0, stream>>>(out, bias, ndD);
        const int wpb = 8;
        edge_score_max<<<(NE + wpb - 1) / wpb, 256, 0, stream>>>(
            fs, fd, a, src, dst, score, segmax, NE);
        const long eh = (long)NE * 4;
        edge_exp_sum<<<(int)((eh + 255) / 256), 256, 0, stream>>>(
            segmax, dst, score, segsum, eh);
        edge_aggregate<<<(NE + wpb - 1) / wpb, 256, 0, stream>>>(
            fs, score, segsum, src, dst, out, NE);
    };

    // ---- input projections ----
    gemm(xp,    WpW_t, bp_word, hw, N_WORD, DIM, 320, 320, 0);
    gemm(x_doc, WpD_t, bp_doc,  hd, N_DOC,  DIM, DIM, DIM, 0);

    // ---- 2 hetero-GAT layers ----
    for (int l = 0; l < 2; ++l) {
        const int r0 = l * 3 + 0, r1 = l * 3 + 1, r2 = l * 3 + 2;
        gat(hw, N_WORD, hd, N_DOC, Wsrc_t[r0], bsrc + r0 * DIM, Wdst_t[r0],
            bdst + r0 * DIM, attn + r0 * DIM, bconv + r0 * DIM, src_wd, dst_wd, owd);
        gat(hd, N_DOC, hw, N_WORD, Wsrc_t[r1], bsrc + r1 * DIM, Wdst_t[r1],
            bdst + r1 * DIM, attn + r1 * DIM, bconv + r1 * DIM, src_dw, dst_dw, odw);
        gat(hd, N_DOC, hd, N_DOC, Wsrc_t[r2], bsrc + r2 * DIM, Wdst_t[r2],
            bdst + r2 * DIM, attn + r2 * DIM, bconv + r2 * DIM, src_dd, dst_dd, odd);
        combine_mean2<<<(int)((ND_D + 255) / 256), 256, 0, stream>>>(hd, owd, odd, ND_D);
        float* t = hw; hw = odw; odw = t;           // h_w <- o_dw
    }

    // ---- MLP head ----
    gemm(hd, Wc1_t, bc1, z, N_DOC, 64, DIM, DIM, 1);
    gemm(z,  Wc2_t, bc2, (float*)d_out, N_DOC, 64, 64, 64, 0);
}
